// Decoder_39479339385520
// MI455X (gfx1250) — matrix operand order repaired and checked by reading, not compiled, not run on hardware
//
#include <hip/hip_runtime.h>
#include <hip/hip_bf16.h>

// ---------------------------------------------------------------------------
// LSTM decoder LM for MI455X (gfx1250, wave32, WMMA).
// All GEMMs: v_wmma_f32_16x16x32_bf16 (bf16 operands, fp32 accumulate) with
// 2x2 register tiling per wave (32x32 output) -> 4 WMMAs per 8 b128 loads.
// bf16 weights (~118 MB) are L2-resident (192 MB L2).
// ---------------------------------------------------------------------------

typedef __attribute__((ext_vector_type(16))) __bf16 bf16x16;
typedef __attribute__((ext_vector_type(8)))  __bf16 bf16x8;
typedef __attribute__((ext_vector_type(8)))  float  f32x8;

union BF16Frag { bf16x16 v; bf16x8 h[2]; };

__device__ __forceinline__ unsigned short f32_to_bf16_bits(float f) {
  unsigned u = __float_as_uint(f);
  u += 0x7FFFu + ((u >> 16) & 1u);          // round-to-nearest-even
  return (unsigned short)(u >> 16);
}

__device__ __forceinline__ float sigf(float x) {
  return 1.0f / (1.0f + __expf(-x));
}

// A-fragment: 16x32 bf16 tile, row m = m0 + lane%16, two contiguous 16B runs.
__device__ __forceinline__ bf16x16 load_a_frag(const unsigned short* row_k, int half) {
  BF16Frag f;
  f.h[0] = *(const bf16x8*)(row_k + half * 8);
  f.h[1] = *(const bf16x8*)(row_k + 16 + half * 8);
  return f.v;
}

// B-fragment: 32x16 tile from row-major W[N,K], row n = n0 + lane%16.
__device__ __forceinline__ bf16x16 load_b_frag(const unsigned short* row_k, int half) {
  BF16Frag f;
  f.h[0] = *(const bf16x8*)(row_k + half * 8);
  f.h[1] = *(const bf16x8*)(row_k + 16 + half * 8);
  return f.v;
}

#define WMMA_BF16(acc, a, b) \
  acc = __builtin_amdgcn_wmma_f32_16x16x32_bf16(false, a, false, b, (short)0, acc, false, false)

// ---------------------------------------------------------------------------
// fp32 -> bf16 conversion (grid-stride)
// ---------------------------------------------------------------------------
__global__ void k_f32_to_bf16(const float* __restrict__ src,
                              unsigned short* __restrict__ dst, size_t n) {
  size_t i = (size_t)blockIdx.x * blockDim.x + threadIdx.x;
  size_t stride = (size_t)gridDim.x * blockDim.x;
  for (; i < n; i += stride) dst[i] = f32_to_bf16_bits(src[i]);
}

__global__ void k_init_state(const float* __restrict__ h0, const float* __restrict__ c0,
                             unsigned short* __restrict__ hbf, float* __restrict__ c, int n) {
  int i = blockIdx.x * blockDim.x + threadIdx.x;
  if (i < n) { hbf[i] = f32_to_bf16_bits(h0[i]); c[i] = c0[i]; }
}

// ---------------------------------------------------------------------------
// Embedding gather + projection: X[m,n] = sum_e emb[tok[m], e] * W_e2h[n, e]
// M = B*S (token-gathered rows), N = H, K = E. 2x2 wave tile. Output bf16.
// ---------------------------------------------------------------------------
__global__ void k_embed_gemm(const unsigned short* __restrict__ emb_bf, // [V,E]
                             const int* __restrict__ tokens,            // [M]
                             const unsigned short* __restrict__ W,      // [N,K]
                             unsigned short* __restrict__ Xbf,          // [M,N]
                             int N, int K) {
  const int lane = threadIdx.x & 31, half = lane >> 4, l16 = lane & 15;
  const int wave = threadIdx.x >> 5;
  const int n0 = (blockIdx.x * 8 + wave) * 32;   // 2 N-tiles per wave
  const int m0 = blockIdx.y * 32;                // 2 M-tiles per wave
  const unsigned short* ar0 = emb_bf + (size_t)tokens[m0 + l16] * K;
  const unsigned short* ar1 = emb_bf + (size_t)tokens[m0 + 16 + l16] * K;
  const unsigned short* wr0 = W + (size_t)(n0 + l16) * K;
  const unsigned short* wr1 = W + (size_t)(n0 + 16 + l16) * K;
  f32x8 acc00 = {}, acc01 = {}, acc10 = {}, acc11 = {};
  for (int k = 0; k < K; k += 32) {
    bf16x16 a0 = load_a_frag(ar0 + k, half);
    bf16x16 a1 = load_a_frag(ar1 + k, half);
    bf16x16 b0 = load_b_frag(wr0 + k, half);
    bf16x16 b1 = load_b_frag(wr1 + k, half);
    WMMA_BF16(acc00, a0, b0); WMMA_BF16(acc01, a0, b1);
    WMMA_BF16(acc10, a1, b0); WMMA_BF16(acc11, a1, b1);
  }
  unsigned short* c0 = Xbf + (size_t)(m0 + half * 8) * N + n0 + l16;
  unsigned short* c1 = Xbf + (size_t)(m0 + 16 + half * 8) * N + n0 + l16;
  for (int r = 0; r < 8; ++r) {
    c0[(size_t)r * N]      = f32_to_bf16_bits(acc00[r]);
    c0[(size_t)r * N + 16] = f32_to_bf16_bits(acc01[r]);
    c1[(size_t)r * N]      = f32_to_bf16_bits(acc10[r]);
    c1[(size_t)r * N + 16] = f32_to_bf16_bits(acc11[r]);
  }
}

// ---------------------------------------------------------------------------
// Gate pre-activations: gates = x_t @ Wih^T + h @ Whh^T + (bih + bhh), fp32.
// M = B = 32 (one 2-tile group), N = 4H, dual K accumulation, 2x2 wave tile.
// ---------------------------------------------------------------------------
__global__ void k_gates_gemm(const unsigned short* __restrict__ xt,   // + b*xstride = x[b,t,:]
                             const unsigned short* __restrict__ hbf,  // [B,H]
                             const unsigned short* __restrict__ Wih,  // [4H,H]
                             const unsigned short* __restrict__ Whh,  // [4H,H]
                             const float* __restrict__ bih, const float* __restrict__ bhh,
                             float* __restrict__ gates,               // [B,4H]
                             int H, int fourH, size_t xstride) {
  const int lane = threadIdx.x & 31, half = lane >> 4, l16 = lane & 15;
  const int wave = threadIdx.x >> 5;
  const int n0 = (blockIdx.x * 8 + wave) * 32;
  const unsigned short* ax0 = xt  + (size_t)l16 * xstride;
  const unsigned short* ax1 = xt  + (size_t)(16 + l16) * xstride;
  const unsigned short* ah0 = hbf + (size_t)l16 * H;
  const unsigned short* ah1 = hbf + (size_t)(16 + l16) * H;
  const unsigned short* wi0 = Wih + (size_t)(n0 + l16) * H;
  const unsigned short* wi1 = Wih + (size_t)(n0 + 16 + l16) * H;
  const unsigned short* wh0 = Whh + (size_t)(n0 + l16) * H;
  const unsigned short* wh1 = Whh + (size_t)(n0 + 16 + l16) * H;
  f32x8 acc00 = {}, acc01 = {}, acc10 = {}, acc11 = {};
  for (int k = 0; k < H; k += 32) {
    bf16x16 a0 = load_a_frag(ax0 + k, half);
    bf16x16 a1 = load_a_frag(ax1 + k, half);
    bf16x16 b0 = load_b_frag(wi0 + k, half);
    bf16x16 b1 = load_b_frag(wi1 + k, half);
    WMMA_BF16(acc00, a0, b0); WMMA_BF16(acc01, a0, b1);
    WMMA_BF16(acc10, a1, b0); WMMA_BF16(acc11, a1, b1);
  }
  for (int k = 0; k < H; k += 32) {
    bf16x16 a0 = load_a_frag(ah0 + k, half);
    bf16x16 a1 = load_a_frag(ah1 + k, half);
    bf16x16 b0 = load_b_frag(wh0 + k, half);
    bf16x16 b1 = load_b_frag(wh1 + k, half);
    WMMA_BF16(acc00, a0, b0); WMMA_BF16(acc01, a0, b1);
    WMMA_BF16(acc10, a1, b0); WMMA_BF16(acc11, a1, b1);
  }
  const float bv0 = bih[n0 + l16] + bhh[n0 + l16];
  const float bv1 = bih[n0 + 16 + l16] + bhh[n0 + 16 + l16];
  float* c0 = gates + (size_t)(half * 8) * fourH + n0 + l16;
  float* c1 = gates + (size_t)(16 + half * 8) * fourH + n0 + l16;
  for (int r = 0; r < 8; ++r) {
    c0[(size_t)r * fourH]      = acc00[r] + bv0;
    c0[(size_t)r * fourH + 16] = acc01[r] + bv1;
    c1[(size_t)r * fourH]      = acc10[r] + bv0;
    c1[(size_t)r * fourH + 16] = acc11[r] + bv1;
  }
}

// ---------------------------------------------------------------------------
// LSTM cell pointwise: c' = sig(f)*c + sig(i)*tanh(g); h' = sig(o)*tanh(c')
// ---------------------------------------------------------------------------
__global__ void k_lstm_cell(const float* __restrict__ gates, float* __restrict__ c,
                            unsigned short* __restrict__ hbf, int H, int n) {
  int i = blockIdx.x * blockDim.x + threadIdx.x;
  if (i >= n) return;
  int b = i / H, j = i - b * H;
  const float* g = gates + (size_t)b * 4 * H;
  float ig = sigf(g[j]);
  float fg = sigf(g[H + j]);
  float gg = tanhf(g[2 * H + j]);
  float og = sigf(g[3 * H + j]);
  float cn = fg * c[i] + ig * gg;
  c[i] = cn;
  hbf[i] = f32_to_bf16_bits(og * tanhf(cn));
}

// ---------------------------------------------------------------------------
// Projection: outs[b,t,:] = tanh(h @ W_proj^T). M=B, N=H, K=H, 2x2 wave tile.
// ---------------------------------------------------------------------------
__global__ void k_proj_gemm(const unsigned short* __restrict__ hbf,  // [B,H]
                            const unsigned short* __restrict__ Wp,   // [H,H]
                            unsigned short* __restrict__ outs_t,     // + b*ostride
                            int H, size_t ostride) {
  const int lane = threadIdx.x & 31, half = lane >> 4, l16 = lane & 15;
  const int wave = threadIdx.x >> 5;
  const int n0 = (blockIdx.x * 8 + wave) * 32;
  const unsigned short* ar0 = hbf + (size_t)l16 * H;
  const unsigned short* ar1 = hbf + (size_t)(16 + l16) * H;
  const unsigned short* wr0 = Wp + (size_t)(n0 + l16) * H;
  const unsigned short* wr1 = Wp + (size_t)(n0 + 16 + l16) * H;
  f32x8 acc00 = {}, acc01 = {}, acc10 = {}, acc11 = {};
  for (int k = 0; k < H; k += 32) {
    bf16x16 a0 = load_a_frag(ar0 + k, half);
    bf16x16 a1 = load_a_frag(ar1 + k, half);
    bf16x16 b0 = load_b_frag(wr0 + k, half);
    bf16x16 b1 = load_b_frag(wr1 + k, half);
    WMMA_BF16(acc00, a0, b0); WMMA_BF16(acc01, a0, b1);
    WMMA_BF16(acc10, a1, b0); WMMA_BF16(acc11, a1, b1);
  }
  unsigned short* c0 = outs_t + (size_t)(half * 8) * ostride + n0 + l16;
  unsigned short* c1 = outs_t + (size_t)(16 + half * 8) * ostride + n0 + l16;
  for (int r = 0; r < 8; ++r) {
    c0[(size_t)r * ostride]      = f32_to_bf16_bits(tanhf(acc00[r]));
    c0[(size_t)r * ostride + 16] = f32_to_bf16_bits(tanhf(acc01[r]));
    c1[(size_t)r * ostride]      = f32_to_bf16_bits(tanhf(acc10[r]));
    c1[(size_t)r * ostride + 16] = f32_to_bf16_bits(tanhf(acc11[r]));
  }
}

// ---------------------------------------------------------------------------
// Logits: C = outs @ W_out^T + b_out. M=2016, N=32000, K=1024. Dominant GEMM.
// 2x2 wave tile; A-rows shared by all 8 waves of a block (L0 hits).
// ---------------------------------------------------------------------------
__global__ void k_out_gemm(const unsigned short* __restrict__ A,   // [M,K]
                           const unsigned short* __restrict__ W,   // [N,K]
                           const float* __restrict__ bias,         // [N]
                           float* __restrict__ C, int N, int K) {
  const int lane = threadIdx.x & 31, half = lane >> 4, l16 = lane & 15;
  const int wave = threadIdx.x >> 5;
  const int n0 = (blockIdx.x * 8 + wave) * 32;
  const int m0 = blockIdx.y * 32;
  const unsigned short* ar0 = A + (size_t)(m0 + l16) * K;
  const unsigned short* ar1 = A + (size_t)(m0 + 16 + l16) * K;
  const unsigned short* wr0 = W + (size_t)(n0 + l16) * K;
  const unsigned short* wr1 = W + (size_t)(n0 + 16 + l16) * K;
  f32x8 acc00 = {}, acc01 = {}, acc10 = {}, acc11 = {};
  for (int k = 0; k < K; k += 32) {
    bf16x16 a0 = load_a_frag(ar0 + k, half);
    bf16x16 a1 = load_a_frag(ar1 + k, half);
    bf16x16 b0 = load_b_frag(wr0 + k, half);
    bf16x16 b1 = load_b_frag(wr1 + k, half);
    WMMA_BF16(acc00, a0, b0); WMMA_BF16(acc01, a0, b1);
    WMMA_BF16(acc10, a1, b0); WMMA_BF16(acc11, a1, b1);
  }
  const float bv0 = bias[n0 + l16];
  const float bv1 = bias[n0 + 16 + l16];
  float* c0 = C + (size_t)(m0 + half * 8) * N + n0 + l16;
  float* c1 = C + (size_t)(m0 + 16 + half * 8) * N + n0 + l16;
  for (int r = 0; r < 8; ++r) {
    c0[(size_t)r * N]      = acc00[r] + bv0;
    c0[(size_t)r * N + 16] = acc01[r] + bv1;
    c1[(size_t)r * N]      = acc10[r] + bv0;
    c1[(size_t)r * N + 16] = acc11[r] + bv1;
  }
}

// ---------------------------------------------------------------------------
// In-place row-wise log_softmax over V (one block per row)
// ---------------------------------------------------------------------------
__global__ void k_log_softmax(float* __restrict__ out, int V) {
  float* p = out + (size_t)blockIdx.x * V;
  __shared__ float red[256];
  const int tid = threadIdx.x;
  float m = -3.402823466e+38f;
  for (int i = tid; i < V; i += 256) m = fmaxf(m, p[i]);
  red[tid] = m; __syncthreads();
  for (int s = 128; s > 0; s >>= 1) {
    if (tid < s) red[tid] = fmaxf(red[tid], red[tid + s]);
    __syncthreads();
  }
  m = red[0]; __syncthreads();
  float sum = 0.f;
  for (int i = tid; i < V; i += 256) sum += __expf(p[i] - m);
  red[tid] = sum; __syncthreads();
  for (int s = 128; s > 0; s >>= 1) {
    if (tid < s) red[tid] += red[tid + s];
    __syncthreads();
  }
  const float lse = m + __logf(red[0]);
  for (int i = tid; i < V; i += 256) p[i] -= lse;
}

// ---------------------------------------------------------------------------
extern "C" void kernel_launch(void* const* d_in, const int* in_sizes, int n_in,
                              void* d_out, int out_size, void* d_ws, size_t ws_size,
                              hipStream_t stream) {
  (void)in_sizes; (void)n_in; (void)out_size; (void)ws_size;
  const int B = 32, S = 64, E = 512, H = 1024, V = 32000;
  const int SM1 = S - 1;

  const float* h0     = (const float*)d_in[0];
  const float* c0     = (const float*)d_in[1];
  const int*   tokens = (const int*)d_in[2];
  // d_in[3] target_len_batch: unused by the reference computation
  const float* emb    = (const float*)d_in[4];
  const float* W_e2h  = (const float*)d_in[5];
  const float* W_ih   = (const float*)d_in[6];
  const float* W_hh   = (const float*)d_in[7];
  const float* b_ih   = (const float*)d_in[8];
  const float* b_hh   = (const float*)d_in[9];
  const float* W_proj = (const float*)d_in[10];
  const float* W_out  = (const float*)d_in[11];
  const float* b_out  = (const float*)d_in[12];
  float* out = (float*)d_out;

  // Workspace bump allocator (256B aligned)
  char* wp = (char*)d_ws;
  auto alloc = [&](size_t bytes) -> char* {
    char* p = wp;
    wp += (bytes + 255) & ~(size_t)255;
    return p;
  };
  unsigned short* emb_bf   = (unsigned short*)alloc((size_t)V * E * 2);
  unsigned short* We2h_bf  = (unsigned short*)alloc((size_t)H * E * 2);
  unsigned short* Wih_bf   = (unsigned short*)alloc((size_t)4 * H * H * 2);
  unsigned short* Whh_bf   = (unsigned short*)alloc((size_t)4 * H * H * 2);
  unsigned short* Wproj_bf = (unsigned short*)alloc((size_t)H * H * 2);
  unsigned short* Wout_bf  = (unsigned short*)alloc((size_t)V * H * 2);
  unsigned short* x_bf     = (unsigned short*)alloc((size_t)B * S * H * 2);   // [B,S,H]
  unsigned short* h_bf     = (unsigned short*)alloc((size_t)B * H * 2);
  float*          c_st     = (float*)alloc((size_t)B * H * 4);
  float*          gates    = (float*)alloc((size_t)B * 4 * H * 4);
  unsigned short* outs_bf  = (unsigned short*)alloc((size_t)B * SM1 * H * 2); // [B,S-1,H]

  // 1) Weight conversion fp32 -> bf16 (one HBM pass; then L2-resident)
  k_f32_to_bf16<<<2048, 256, 0, stream>>>(emb,    emb_bf,   (size_t)V * E);
  k_f32_to_bf16<<<512,  256, 0, stream>>>(W_e2h,  We2h_bf,  (size_t)H * E);
  k_f32_to_bf16<<<2048, 256, 0, stream>>>(W_ih,   Wih_bf,   (size_t)4 * H * H);
  k_f32_to_bf16<<<2048, 256, 0, stream>>>(W_hh,   Whh_bf,   (size_t)4 * H * H);
  k_f32_to_bf16<<<1024, 256, 0, stream>>>(W_proj, Wproj_bf, (size_t)H * H);
  k_f32_to_bf16<<<4096, 256, 0, stream>>>(W_out,  Wout_bf,  (size_t)V * H);
  k_init_state<<<(B * H + 255) / 256, 256, 0, stream>>>(h0, c0, h_bf, c_st, B * H);

  // 2) Embedding gather + projection: [B*S, E] x [E, H] -> x_bf [B,S,H]
  //    wave tile 32x32; block covers 32 rows x 256 cols; grid (1024/256, 2048/32)
  k_embed_gemm<<<dim3(H / 256, (B * S) / 32), 256, 0, stream>>>(
      emb_bf, tokens, We2h_bf, x_bf, H, E);

  // 3) Sequential LSTM recurrence (63 steps; latency-critical)
  for (int t = 0; t < SM1; ++t) {
    k_gates_gemm<<<dim3(4 * H / 256, 1), 256, 0, stream>>>(
        x_bf + (size_t)t * H, h_bf, Wih_bf, Whh_bf, b_ih, b_hh,
        gates, H, 4 * H, (size_t)S * H);
    k_lstm_cell<<<(B * H + 255) / 256, 256, 0, stream>>>(gates, c_st, h_bf, H, B * H);
    k_proj_gemm<<<dim3(H / 256, 1), 256, 0, stream>>>(
        h_bf, Wproj_bf, outs_bf + (size_t)t * H, H, (size_t)SM1 * H);
  }

  // 4) Logits GEMM (dominant): [2016,1024] x [1024,32000] + bias
  //    grid (32000/256, 2016/32) = (125, 63)
  k_out_gemm<<<dim3(V / 256, (B * SM1) / 32), 256, 0, stream>>>(
      outs_bf, Wout_bf, b_out, out, V, H);

  // 5) Row-wise log_softmax in place on d_out
  k_log_softmax<<<B * SM1, 256, 0, stream>>>(out, V);
}
